// WindowAttention_57148834841122
// MI455X (gfx1250) — compile-verified, matrix operand order checked by reading, not run on hardware
//
#include <hip/hip_runtime.h>

#define NTOK 25
#define DIM  128
#define NH   4
#define WSQ  49
#define XSTR 136   // halves stride for 128-wide LDS tiles (16B aligned, bank-skewed)
#define HSTR 40    // halves stride for 32-wide LDS tiles
#define BMSTR 36

typedef __attribute__((ext_vector_type(16))) __bf16 v16bf;
typedef __attribute__((ext_vector_type(8)))  __bf16 v8bf;
typedef __attribute__((ext_vector_type(2)))  __bf16 v2bf;
typedef __attribute__((ext_vector_type(2)))  float  v2f;
typedef __attribute__((ext_vector_type(8)))  float  v8f;

// Packed f32x2 -> bf16x2 (RNE) expressed as a vector fptrunc so the backend can
// select a native packed convert if gfx1250 has one.
__device__ __forceinline__ unsigned pk_bf16(float a, float b) {
  v2f t; t.x = a; t.y = b;
  union { v2bf v; unsigned u; } c;
  c.v = __builtin_convertvector(t, v2bf);
  return c.u;
}

__device__ __forceinline__ unsigned short f2bf(float f) {
  union { __bf16 b; unsigned short s; } c;
  c.b = (__bf16)f;
  return c.s;
}

// Load a 16x32 bf16 fragment (A-operand layout; also used for B via N x K row-major
// storage). Lane l -> row r0 + (l&15); lanes 0-15 take K chunks {0-7,16-23},
// lanes 16-31 take {8-15,24-31}. Two 16B loads.
__device__ __forceinline__ v16bf load_frag(const unsigned short* base, int stride_h,
                                           int r0, int k0) {
  const int l = threadIdx.x & 31;
  const unsigned short* p = base + (r0 + (l & 15)) * stride_h + k0 + ((l >> 4) << 3);
  union { v16bf v; v8bf h[2]; } u;
  u.h[0] = *(const v8bf*)(p);
  u.h[1] = *(const v8bf*)(p + 16);
  return u.v;
}

__device__ __forceinline__ v8f wmma_bf16(v16bf a, v16bf b, v8f c) {
  return __builtin_amdgcn_wmma_f32_16x16x32_bf16(false, a, false, b, (short)0, c,
                                                 false, false);
}

// Preprocess: f32 -> bf16 weights into workspace; fold 1/sqrt(hd) into Q rows.
__global__ void prep_kernel(const float* __restrict__ qkv_w, const float* __restrict__ qkv_b,
                            const float* __restrict__ proj_w, const float* __restrict__ proj_b,
                            unsigned short* __restrict__ wq, unsigned short* __restrict__ wp,
                            float* __restrict__ bq, float* __restrict__ bp) {
  const float scale = 0.17677669529663687f; // 32^-0.5
  int i = blockIdx.x * blockDim.x + threadIdx.x;
  if (i < 49152) {
    int j = i >> 7; // output feature row of qkv_w (384x128)
    wq[i] = f2bf(qkv_w[i] * (j < 128 ? scale : 1.0f));
  } else if (i < 65536) {
    int k = i - 49152;
    wp[k] = f2bf(proj_w[k]);
  } else if (i < 65920) {
    int j = i - 65536;
    bq[j] = qkv_b[j] * (j < 128 ? scale : 1.0f);
  } else if (i < 66048) {
    int j = i - 65920;
    bp[j] = proj_b[j];
  }
}

__global__ __launch_bounds__(256) void win_attn_kernel(
    const float* __restrict__ x, const float* __restrict__ mask,
    const int* __restrict__ ids_keep, const float* __restrict__ bias_table,
    const int* __restrict__ rel_index,
    const unsigned short* __restrict__ wq, const unsigned short* __restrict__ wp,
    const float* __restrict__ bq, const float* __restrict__ bp,
    float* __restrict__ out) {
  __shared__ __align__(16) unsigned short sX[32 * XSTR];
  __shared__ __align__(16) unsigned short sO[32 * XSTR];
  __shared__ __align__(16) unsigned short sQ[NH][32 * HSTR];
  __shared__ __align__(16) unsigned short sK[NH][32 * HSTR];
  __shared__ __align__(16) unsigned short sVt[NH][32 * HSTR];
  __shared__ __align__(16) unsigned short sP[NH][32 * HSTR];
  __shared__ float sBM[NH][32 * BMSTR];
  __shared__ int sIds[32];

  const int b   = blockIdx.x;
  const int w   = b & 63;          // BS == NW == 64, both gathers index by b % 64
  const int tid = threadIdx.x;
  const int l   = tid & 31;
  const int wid = tid >> 5;        // 8 waves

  if (tid < NTOK) sIds[tid] = ids_keep[w * NTOK + tid];
  __syncthreads();

  // ---- Stage 1: x -> bf16 LDS via packed converts (pad rows 25..31 with zeros) ----
  {
    const int row = tid >> 3;
    const int cb  = (tid & 7) << 4;
    uint4* dst = (uint4*)(sX + row * XSTR + cb);
    if (row < NTOK) {
      const float4* src = (const float4*)(x + ((size_t)b * NTOK + row) * DIM + cb);
      float4 v0 = src[0], v1 = src[1], v2 = src[2], v3 = src[3];
      dst[0] = make_uint4(pk_bf16(v0.x, v0.y), pk_bf16(v0.z, v0.w),
                          pk_bf16(v1.x, v1.y), pk_bf16(v1.z, v1.w));
      dst[1] = make_uint4(pk_bf16(v2.x, v2.y), pk_bf16(v2.z, v2.w),
                          pk_bf16(v3.x, v3.y), pk_bf16(v3.z, v3.w));
    } else {
      dst[0] = make_uint4(0, 0, 0, 0);
      dst[1] = make_uint4(0, 0, 0, 0);
    }
  }

  // ---- Stage 0b: fused (relative-position bias + attention mask) gather ----
  for (int e = tid; e < NH * NTOK * NTOK; e += 256) {
    const int h   = e / (NTOK * NTOK);
    const int rem = e - h * NTOK * NTOK;
    const int i   = rem / NTOK;
    const int j   = rem - i * NTOK;
    const int off = sIds[i] * WSQ + sIds[j];
    const int idx = rel_index[off];
    sBM[h][i * BMSTR + j] = bias_table[idx * NH + h] + mask[(size_t)w * (WSQ * WSQ) + off];
  }
  __syncthreads();

  // ---- Stage 2: QKV = X @ Wqkv^T + b (bf16 WMMA; B-frags hoisted across M-tiles) ----
#pragma unroll
  for (int t = 0; t < 3; ++t) {
    const int nt = wid * 3 + t;                  // 0..23 output-feature tile
    const float bval = bq[nt * 16 + (l & 15)];
    v16bf B0 = load_frag(wq, DIM, nt * 16, 0);
    v16bf B1 = load_frag(wq, DIM, nt * 16, 32);
    v16bf B2 = load_frag(wq, DIM, nt * 16, 64);
    v16bf B3 = load_frag(wq, DIM, nt * 16, 96);
#pragma unroll
    for (int mt = 0; mt < 2; ++mt) {
      v8f acc = {};
      acc = wmma_bf16(load_frag(sX, XSTR, mt * 16, 0),  B0, acc);
      acc = wmma_bf16(load_frag(sX, XSTR, mt * 16, 32), B1, acc);
      acc = wmma_bf16(load_frag(sX, XSTR, mt * 16, 64), B2, acc);
      acc = wmma_bf16(load_frag(sX, XSTR, mt * 16, 96), B3, acc);

      const int head = (nt & 7) >> 1;
      const int cloc = (nt & 1) << 4;
      const int col  = cloc + (l & 15);
      const int mb   = mt * 16 + ((l >> 4) << 3);
      if (nt < 8) {            // Q (scale folded in weights)
        unsigned short* dst = sQ[head];
#pragma unroll
        for (int r = 0; r < 8; ++r)
          dst[(mb + r) * HSTR + col] = f2bf(acc[r] + bval);
      } else if (nt < 16) {    // K
        unsigned short* dst = sK[head];
#pragma unroll
        for (int r = 0; r < 8; ++r)
          dst[(mb + r) * HSTR + col] = f2bf(acc[r] + bval);
      } else {                 // V stored transposed (hd x tokens) for B-operand
        unsigned* p32 = (unsigned*)(sVt[head] + col * HSTR + mb);
#pragma unroll
        for (int r = 0; r < 8; r += 2)
          p32[r >> 1] = pk_bf16(acc[r] + bval, acc[r + 1] + bval);
      }
    }
  }
  __syncthreads();

  // ---- Stage 3: per-(head, M-tile) attention: S = Q@K^T, softmax, O = P@V ----
  {
    const int head = wid >> 1;
    const int mt   = wid & 1;
    v8f z = {};
    v16bf qa  = load_frag(sQ[head], HSTR, mt * 16, 0);
    v16bf kb0 = load_frag(sK[head], HSTR, 0, 0);
    v16bf kb1 = load_frag(sK[head], HSTR, 16, 0);
    v8f s0 = wmma_bf16(qa, kb0, z);
    v8f s1 = wmma_bf16(qa, kb1, z);

    const int c0 = l & 15;
    const bool v1 = (c0 < 9);                    // col 16+c0 < 25
    const int mb = mt * 16 + ((l >> 4) << 3);
#pragma unroll
    for (int r = 0; r < 8; ++r) {
      const int m = mb + r;
      float a0 = s0[r] + sBM[head][m * BMSTR + c0];
      float a1 = v1 ? (s1[r] + sBM[head][m * BMSTR + 16 + c0]) : -3.0e38f;
      float vm = fmaxf(a0, a1);
#pragma unroll
      for (int off = 8; off >= 1; off >>= 1)      // reduce within 16-lane halves
        vm = fmaxf(vm, __shfl_xor(vm, off, 32));
      float e0 = __expf(a0 - vm);
      float e1 = v1 ? __expf(a1 - vm) : 0.0f;
      float sum = e0 + e1;
#pragma unroll
      for (int off = 8; off >= 1; off >>= 1)
        sum += __shfl_xor(sum, off, 32);
      float inv = __builtin_amdgcn_rcpf(sum);    // single v_rcp_f32
      unsigned pp = pk_bf16(e0 * inv, e1 * inv);
      sP[head][m * HSTR + c0]      = (unsigned short)pp;
      sP[head][m * HSTR + 16 + c0] = (unsigned short)(pp >> 16);
    }

    v16bf pa  = load_frag(sP[head],  HSTR, mt * 16, 0);
    v16bf vb0 = load_frag(sVt[head], HSTR, 0, 0);
    v16bf vb1 = load_frag(sVt[head], HSTR, 16, 0);
    v8f o0 = wmma_bf16(pa, vb0, z);
    v8f o1 = wmma_bf16(pa, vb1, z);

    const int col0 = head * 32 + c0;
#pragma unroll
    for (int r = 0; r < 8; ++r) {
      unsigned po = pk_bf16(o0[r], o1[r]);
      sO[(mb + r) * XSTR + col0]      = (unsigned short)po;
      sO[(mb + r) * XSTR + col0 + 16] = (unsigned short)(po >> 16);
    }
  }
  __syncthreads();

  // ---- Stage 4: out = O @ proj_w^T + proj_b ----
#pragma unroll
  for (int t = 0; t < 2; ++t) {
    const int tt = wid * 2 + t;                  // 0..15
    const int mt = tt >> 3;
    const int nt = tt & 7;
    v8f acc = {};
#pragma unroll
    for (int ks = 0; ks < 4; ++ks) {
      v16bf a  = load_frag(sO, XSTR, mt * 16, ks * 32);
      v16bf bb = load_frag(wp, DIM, nt * 16, ks * 32);
      acc = wmma_bf16(a, bb, acc);
    }
    const int col = nt * 16 + (l & 15);
    const float pbv = bp[col];
    const int mb = mt * 16 + ((l >> 4) << 3);
    float* obase = out + (size_t)b * NTOK * DIM;
#pragma unroll
    for (int r = 0; r < 8; ++r) {
      const int row = mb + r;
      if (row < NTOK) obase[row * DIM + col] = acc[r] + pbv;
    }
  }
}

extern "C" void kernel_launch(void* const* d_in, const int* in_sizes, int n_in,
                              void* d_out, int out_size, void* d_ws, size_t ws_size,
                              hipStream_t stream) {
  const float* x          = (const float*)d_in[0];
  const float* mask       = (const float*)d_in[1];
  const int*   ids_keep   = (const int*)d_in[2];
  const float* qkv_w      = (const float*)d_in[3];
  const float* qkv_b      = (const float*)d_in[4];
  const float* proj_w     = (const float*)d_in[5];
  const float* proj_b     = (const float*)d_in[6];
  const float* bias_table = (const float*)d_in[7];
  const int*   rel_index  = (const int*)d_in[8];

  // Workspace layout: bf16 qkv_w (49152 halves) | bf16 proj_w (16384 halves) |
  //                   f32 qkv_b (384, Q-scaled) | f32 proj_b (128)
  unsigned short* wq = (unsigned short*)d_ws;
  unsigned short* wp = wq + 49152;
  float* bqp = (float*)(wp + 16384);
  float* bpp = bqp + 384;

  prep_kernel<<<259, 256, 0, stream>>>(qkv_w, qkv_b, proj_w, proj_b, wq, wp, bqp, bpp);

  const int nWin = in_sizes[0] / (NTOK * DIM);   // 4096
  win_attn_kernel<<<nWin, 256, 0, stream>>>(x, mask, ids_keep, bias_table, rel_index,
                                            wq, wp, bqp, bpp, (float*)d_out);
}
